// Flux2Attention_35914516529916
// MI455X (gfx1250) — matrix operand order repaired and checked by reading, not compiled, not run on hardware
//
#include <hip/hip_runtime.h>
#include <hip/hip_bf16.h>

// ---------------------------------------------------------------------------
// Types / helpers
// ---------------------------------------------------------------------------
typedef __attribute__((ext_vector_type(16))) __bf16        v16bf;
typedef __attribute__((ext_vector_type(8)))  float         v8f;
typedef __attribute__((ext_vector_type(4)))  float         f32x4;
typedef __attribute__((ext_vector_type(4)))  unsigned int  u32x4;
typedef __attribute__((ext_vector_type(4)))  unsigned short u16x4;

__device__ __forceinline__ unsigned short f2bf(float f) {
  union { float f; unsigned u; } v; v.f = f;
  unsigned u = v.u + 0x7FFFu + ((v.u >> 16) & 1u);   // round-to-nearest-even
  return (unsigned short)(u >> 16);
}

union BF16Frag { v16bf v; u32x4 q[2]; };
union F32Frag  { v8f   v; float f[8]; };
union U16x8    { u32x4 q; unsigned short s[8]; };

__device__ __forceinline__ v8f wmma_bf16(v16bf a, v16bf b, v8f c) {
  return __builtin_amdgcn_wmma_f32_16x16x32_bf16(
      /*neg_a=*/false, a, /*neg_b=*/false, b,
      /*c_mod=*/(short)0, c, /*reuse_a=*/false, /*reuse_b=*/false);
}

// CDNA5 async copy: 16 bytes global -> LDS, tracked by ASYNCcnt.
__device__ __forceinline__ void async_copy16(const void* g, void* l) {
  asm volatile("global_load_async_to_lds_b128 %0, %1, off"
               :: "v"((unsigned)(unsigned long long)l), "v"(g)
               : "memory");
}
__device__ __forceinline__ void wait_async_le3() {
  asm volatile("s_wait_asynccnt 3" ::: "memory");
}
__device__ __forceinline__ void wait_async_0() {
  asm volatile("s_wait_asynccnt 0" ::: "memory");
}

// A fragment (16x32 bf16, row-major, leading dim ld elements, ld % 8 == 0):
// lane%16 = M row; lanes 0-15 hold K {0..7,16..23}, lanes 16-31 K {8..15,24..31}
// -> two contiguous 16B chunks per lane.
__device__ __forceinline__ v16bf load_a_frag(const unsigned short* base, int ld, int lane) {
  const int m  = lane & 15;
  const int kb = (lane < 16) ? 0 : 8;
  BF16Frag f;
  f.q[0] = *(const u32x4*)(base + m * ld + kb);
  f.q[1] = *(const u32x4*)(base + m * ld + kb + 16);
  return f.v;
}

// B fragment from k-contiguous storage src[n][k] (leading dim ld, ld % 8 == 0):
// lane%16 = N col; lanes 0-15 hold K {0..7,16..23}, lanes 16-31 K {8..15,24..31}
// -> two contiguous 16B chunks per lane.
__device__ __forceinline__ v16bf load_bk_frag(const unsigned short* base, int ld, int lane) {
  const int n  = lane & 15;
  const int kb = (lane < 16) ? 0 : 8;
  BF16Frag f;
  f.q[0] = *(const u32x4*)(base + n * ld + kb);
  f.q[1] = *(const u32x4*)(base + n * ld + kb + 16);
  return f.v;
}

// ---------------------------------------------------------------------------
// fp32 -> bf16 cast, 4 elements / thread (n % 4 == 0)
// ---------------------------------------------------------------------------
__global__ void cast_bf16_kernel(const float* __restrict__ src,
                                 unsigned short* __restrict__ dst, long n4) {
  long i = (long)blockIdx.x * blockDim.x + threadIdx.x;
  const long stride = (long)gridDim.x * blockDim.x;
  for (; i < n4; i += stride) {
    f32x4 x = ((const f32x4*)src)[i];
    u16x4 y;
    y.x = f2bf(x.x); y.y = f2bf(x.y); y.z = f2bf(x.z); y.w = f2bf(x.w);
    ((u16x4*)dst)[i] = y;
  }
}

// ---------------------------------------------------------------------------
// fp32 [K][N] -> bf16 [N][K] cast+transpose (coalesced both sides)
// ---------------------------------------------------------------------------
__global__ __launch_bounds__(256)
void cast_bf16_t_kernel(const float* __restrict__ W,
                        unsigned short* __restrict__ Wt, int K, int N) {
  __shared__ float tile[32][33];
  const int tx = threadIdx.x, ty = threadIdx.y;   // 32 x 8
  const int k0 = blockIdx.y * 32, n0 = blockIdx.x * 32;
#pragma unroll
  for (int i = 0; i < 4; ++i)
    tile[ty + i * 8][tx] = W[(size_t)(k0 + ty + i * 8) * N + n0 + tx];
  __syncthreads();
#pragma unroll
  for (int i = 0; i < 4; ++i)
    Wt[(size_t)(n0 + ty + i * 8) * K + k0 + tx] = f2bf(tile[tx][ty + i * 8]);
}

// ---------------------------------------------------------------------------
// Tiled WMMA GEMM: C[M,N] = A[M,K](bf16) @ Wt[N,K](bf16, pre-transposed) + bias
// Block tile 64x128, 8 waves x (32x32 patch).  Double-buffered LDS tiles
// filled with global_load_async_to_lds_b128 (3 async instr / wave / tile).
// ---------------------------------------------------------------------------
__global__ __launch_bounds__(256)
void wmma_gemm_kernel(const unsigned short* __restrict__ A,
                      const unsigned short* __restrict__ Bt,
                      const float* __restrict__ bias,
                      float* __restrict__ C,
                      int M, int N, int K) {
  constexpr int LDT = 40;  // padded stride (elements), 80 B = 5 * 16 B
  __shared__ __attribute__((aligned(16))) unsigned short As[2][64 * LDT];
  __shared__ __attribute__((aligned(16))) unsigned short Bs[2][128 * LDT];

  const int tid  = threadIdx.x;
  const int lane = tid & 31;
  const int wave = tid >> 5;
  const int m0   = blockIdx.y * 64;
  const int n0   = blockIdx.x * 128;
  const int wm   = (wave >> 2) * 32;
  const int wn   = (wave & 3) * 32;

  auto stage = [&](int buf, int k0) {
    {  // A tile: 64 rows x 4 chunks of 8 bf16 -> 256 transfers, 1/thread
      const int row = tid >> 2, ck = tid & 3;
      async_copy16(&A[(size_t)(m0 + row) * K + k0 + ck * 8],
                   &As[buf][row * LDT + ck * 8]);
    }
#pragma unroll
    for (int i = 0; i < 2; ++i) {  // B tile: 128 rows x 4 chunks -> 2/thread
      const int idx = i * 256 + tid;
      const int row = idx >> 2, ck = idx & 3;
      async_copy16(&Bt[(size_t)(n0 + row) * K + k0 + ck * 8],
                   &Bs[buf][row * LDT + ck * 8]);
    }
  };

  v8f zero = {};
  v8f acc[2][2];
#pragma unroll
  for (int mi = 0; mi < 2; ++mi)
#pragma unroll
    for (int ni = 0; ni < 2; ++ni) acc[mi][ni] = zero;

  stage(0, 0);
  const int nk = K / 32;
  for (int t = 0; t < nk; ++t) {
    const int buf = t & 1;
    if (t + 1 < nk) {
      stage(buf ^ 1, (t + 1) * 32);
      wait_async_le3();   // drain current tile's 3 async ops; next 3 in flight
    } else {
      wait_async_0();
    }
    __syncthreads();

#pragma unroll
    for (int mi = 0; mi < 2; ++mi) {
      v16bf a = load_a_frag(&As[buf][(wm + mi * 16) * LDT], LDT, lane);
#pragma unroll
      for (int ni = 0; ni < 2; ++ni) {
        v16bf b = load_bk_frag(&Bs[buf][(wn + ni * 16) * LDT], LDT, lane);
        acc[mi][ni] = wmma_bf16(a, b, acc[mi][ni]);
      }
    }
    __syncthreads();
  }

  const int n = lane & 15, half = lane >> 4;
#pragma unroll
  for (int mi = 0; mi < 2; ++mi)
#pragma unroll
    for (int ni = 0; ni < 2; ++ni) {
      const int col = n0 + wn + ni * 16 + n;
      const float bv = bias ? bias[col] : 0.0f;
      F32Frag f; f.v = acc[mi][ni];
#pragma unroll
      for (int r = 0; r < 8; ++r) {
        const int row = m0 + wm + mi * 16 + r + 8 * half;
        C[(size_t)row * N + col] = f.f[r] + bv;
      }
    }
}

// ---------------------------------------------------------------------------
// Fused per-head RMSNorm + RoPE + bf16 pack.  X,Y: [S, H*128].
// ---------------------------------------------------------------------------
__global__ __launch_bounds__(128)
void rmsnorm_rope_kernel(const float* __restrict__ X,
                         const float* __restrict__ w_img,
                         const float* __restrict__ w_enc,
                         const float* __restrict__ rope_cos,
                         const float* __restrict__ rope_sin,
                         unsigned short* __restrict__ Y,
                         int S_txt) {
  const int h = blockIdx.x, s = blockIdx.y, d = threadIdx.x;
  __shared__ float red[128];
  __shared__ float xs[128];

  const size_t idx = (size_t)s * 3072 + h * 128 + d;
  const float x = X[idx];
  red[d] = x * x;
  __syncthreads();
#pragma unroll
  for (int off = 64; off > 0; off >>= 1) {
    if (d < off) red[d] += red[d + off];
    __syncthreads();
  }
  const float inv = rsqrtf(red[0] * (1.0f / 128.0f) + 1e-5f);
  const float* wp = (s < S_txt) ? w_enc : w_img;
  const float xn = x * inv * wp[d];
  xs[d] = xn;
  __syncthreads();
  const float rot = (d & 1) ? xs[d - 1] : -xs[d + 1];
  const float c  = rope_cos[(size_t)s * 128 + d];
  const float sn = rope_sin[(size_t)s * 128 + d];
  Y[idx] = f2bf(xn * c + rot * sn);
}

// ---------------------------------------------------------------------------
// Flash attention: one block per (32-query tile, head).  Online softmax.
// Q,K tiles staged with async-to-LDS; V transposed at staging so every
// fragment load is two aligned 16B LDS reads.
// ---------------------------------------------------------------------------
__global__ __launch_bounds__(256)
void flash_attn_kernel(const unsigned short* __restrict__ Qb,
                       const unsigned short* __restrict__ Kb,
                       const unsigned short* __restrict__ Vb,
                       float* __restrict__ O,
                       int S, int H) {
  constexpr int LDV = 40;
  const int h    = blockIdx.y;
  const int q0   = blockIdx.x * 32;
  const int tid  = threadIdx.x;
  const int lane = tid & 31;
  const int wave = tid >> 5;
  const int D    = H * 128;

  __shared__ __attribute__((aligned(16))) unsigned short Qs[32 * 128];  // 8 KB
  __shared__ __attribute__((aligned(16))) unsigned short Ks[32 * 128];  // 8 KB
  __shared__ __attribute__((aligned(16))) unsigned short Vs[128 * LDV]; // 10 KB  V^T [d][key]
  __shared__ __attribute__((aligned(16))) unsigned short Ps[32 * 32];   // 2 KB
  __shared__ float Ss[32 * 32];                                         // 4 KB
  __shared__ float Os[32 * 128];                                        // 16 KB
  __shared__ float row_m[32], row_l[32], row_a[32];

  // Q tile: async direct copy (2 x b128 / thread)
#pragma unroll
  for (int i = 0; i < 2; ++i) {
    const int idx = i * 256 + tid;
    const int row = idx >> 4, ck = idx & 15;
    async_copy16(&Qb[(size_t)(q0 + row) * D + h * 128 + ck * 8],
                 &Qs[row * 128 + ck * 8]);
  }
#pragma unroll
  for (int i = 0; i < 16; ++i) Os[i * 256 + tid] = 0.0f;
  if (tid < 32) { row_m[tid] = -1e30f; row_l[tid] = 0.0f; }

  const float scale = 0.08838834764831845f;  // 1/sqrt(128)

  for (int k0 = 0; k0 < S; k0 += 32) {
    // K tile: async direct copy
#pragma unroll
    for (int i = 0; i < 2; ++i) {
      const int idx = i * 256 + tid;
      const int row = idx >> 4, ck = idx & 15;
      async_copy16(&Kb[(size_t)(k0 + row) * D + h * 128 + ck * 8],
                   &Ks[row * 128 + ck * 8]);
    }
    // V tile: coalesced 16B global read, transpose-scatter into Vs[d][key]
#pragma unroll
    for (int i = 0; i < 2; ++i) {
      const int idx = i * 256 + tid;
      const int key = idx >> 4, ck = idx & 15;
      U16x8 val;
      val.q = *(const u32x4*)&Vb[(size_t)(k0 + key) * D + h * 128 + ck * 8];
#pragma unroll
      for (int j = 0; j < 8; ++j)
        Vs[(ck * 8 + j) * LDV + key] = val.s[j];
    }
    wait_async_0();
    __syncthreads();

    // ---- scores = Q (32x128) * K^T; waves 0..3 own 16x16 tiles
    if (wave < 4) {
      const int mt = wave & 1, nt = wave >> 1;
      v8f c = {};
#pragma unroll
      for (int kk = 0; kk < 4; ++kk) {
        v16bf a = load_a_frag (&Qs[(mt * 16) * 128 + kk * 32], 128, lane);
        v16bf b = load_bk_frag(&Ks[(nt * 16) * 128 + kk * 32], 128, lane);
        c = wmma_bf16(a, b, c);
      }
      const int n = lane & 15, half = lane >> 4;
      F32Frag f; f.v = c;
#pragma unroll
      for (int r = 0; r < 8; ++r)
        Ss[(mt * 16 + r + 8 * half) * 32 + nt * 16 + n] = f.f[r];
    }
    __syncthreads();

    // ---- online softmax row update
    if (tid < 32) {
      const float m_old = row_m[tid];
      float mx = m_old;
#pragma unroll
      for (int j = 0; j < 32; ++j) mx = fmaxf(mx, Ss[tid * 32 + j] * scale);
      const float alpha = __expf(m_old - mx);
      float psum = 0.0f;
#pragma unroll
      for (int j = 0; j < 32; ++j) {
        const float p = __expf(Ss[tid * 32 + j] * scale - mx);
        psum += p;
        Ps[tid * 32 + j] = f2bf(p);
      }
      row_l[tid] = row_l[tid] * alpha + psum;
      row_m[tid] = mx;
      row_a[tid] = alpha;
    }
    __syncthreads();

    // ---- rescale running output
#pragma unroll
    for (int i = 0; i < 16; ++i) {
      const int e = i * 256 + tid;
      Os[e] *= row_a[e >> 7];
    }
    __syncthreads();

    // ---- O += P (32x32) * V (32x128); 16 tiles, 2 per wave
#pragma unroll
    for (int tt = 0; tt < 2; ++tt) {
      const int t  = wave + tt * 8;
      const int mt = t >> 3, nt = t & 7;
      const int n = lane & 15, half = lane >> 4;
      F32Frag cf;
#pragma unroll
      for (int r = 0; r < 8; ++r)
        cf.f[r] = Os[(mt * 16 + r + 8 * half) * 128 + nt * 16 + n];
      v16bf a = load_a_frag (&Ps[(mt * 16) * 32], 32, lane);
      v16bf b = load_bk_frag(&Vs[(nt * 16) * LDV], LDV, lane);
      cf.v = wmma_bf16(a, b, cf.v);
#pragma unroll
      for (int r = 0; r < 8; ++r)
        Os[(mt * 16 + r + 8 * half) * 128 + nt * 16 + n] = cf.f[r];
    }
    __syncthreads();
  }

  // ---- writeback, normalize by row sum
#pragma unroll
  for (int i = 0; i < 16; ++i) {
    const int e   = i * 256 + tid;
    const int row = e >> 7, d = e & 127;
    O[(size_t)(q0 + row) * D + h * 128 + d] = Os[e] / row_l[row];
  }
}

// ---------------------------------------------------------------------------
// Host-side orchestration
// ---------------------------------------------------------------------------
extern "C" void kernel_launch(void* const* d_in, const int* in_sizes, int n_in,
                              void* d_out, int out_size, void* d_ws, size_t ws_size,
                              hipStream_t stream) {
  (void)in_sizes; (void)n_in; (void)out_size; (void)ws_size;

  const float* hs   = (const float*)d_in[0];
  const float* enc  = (const float*)d_in[1];
  const float* cosb = (const float*)d_in[2];
  const float* sinb = (const float*)d_in[3];
  const float* Wq   = (const float*)d_in[4];
  const float* Wk   = (const float*)d_in[5];
  const float* Wv   = (const float*)d_in[6];
  const float* nqw  = (const float*)d_in[7];
  const float* nkw  = (const float*)d_in[8];
  const float* Wqa  = (const float*)d_in[9];
  const float* bqa  = (const float*)d_in[10];
  const float* Wka  = (const float*)d_in[11];
  const float* bka  = (const float*)d_in[12];
  const float* Wva  = (const float*)d_in[13];
  const float* bva  = (const float*)d_in[14];
  const float* naq  = (const float*)d_in[15];
  const float* nak  = (const float*)d_in[16];
  const float* Wo   = (const float*)d_in[17];
  const float* bo   = (const float*)d_in[18];
  const float* Wao  = (const float*)d_in[19];
  const float* bao  = (const float*)d_in[20];
  float* out = (float*)d_out;

  constexpr int S_IMG = 2048, S_TXT = 512, S = 2560, D = 3072, H = 24;

  char* ws = (char*)d_ws;
  size_t off = 0;
  auto alloc = [&](size_t bytes) -> void* {
    void* p = ws + off;
    off += (bytes + 255) & ~(size_t)255;
    return p;
  };

  unsigned short* hsb  = (unsigned short*)alloc((size_t)S_IMG * D * 2);
  unsigned short* encb = (unsigned short*)alloc((size_t)S_TXT * D * 2);
  unsigned short* Wqb  = (unsigned short*)alloc((size_t)D * D * 2);  // all W*b are
  unsigned short* Wkb  = (unsigned short*)alloc((size_t)D * D * 2);  // TRANSPOSED
  unsigned short* Wvb  = (unsigned short*)alloc((size_t)D * D * 2);  // [N][K] bf16
  unsigned short* Wqab = (unsigned short*)alloc((size_t)D * D * 2);
  unsigned short* Wkab = (unsigned short*)alloc((size_t)D * D * 2);
  unsigned short* Wvab = (unsigned short*)alloc((size_t)D * D * 2);
  unsigned short* Wob  = (unsigned short*)alloc((size_t)D * D * 2);
  unsigned short* Waob = (unsigned short*)alloc((size_t)D * D * 2);
  float* Qf = (float*)alloc((size_t)S * D * 4);
  float* Kf = (float*)alloc((size_t)S * D * 4);
  float* Vf = (float*)alloc((size_t)S * D * 4);
  unsigned short* Qbb = (unsigned short*)alloc((size_t)S * D * 2);
  unsigned short* Kbb = (unsigned short*)alloc((size_t)S * D * 2);
  unsigned short* Vbb = (unsigned short*)alloc((size_t)S * D * 2);
  float* Of = (float*)alloc((size_t)S * D * 4);
  unsigned short* Ob = (unsigned short*)alloc((size_t)S * D * 2);

  auto cast = [&](const float* src, unsigned short* dst, long n) {
    cast_bf16_kernel<<<1024, 256, 0, stream>>>(src, dst, n / 4);
  };
  auto castT = [&](const float* W, unsigned short* Wt) {
    dim3 grid(D / 32, D / 32), block(32, 8);
    cast_bf16_t_kernel<<<grid, block, 0, stream>>>(W, Wt, D, D);
  };
  auto gemm = [&](const unsigned short* A, const unsigned short* Bt,
                  const float* bias, float* C, int M) {
    dim3 grid(D / 128, M / 64);
    wmma_gemm_kernel<<<grid, 256, 0, stream>>>(A, Bt, bias, C, M, D, D);
  };

  // 1) bf16 staging (weights cast + transposed once)
  cast(hs,  hsb,  (long)S_IMG * D);
  cast(enc, encb, (long)S_TXT * D);
  castT(Wq,  Wqb);   castT(Wk,  Wkb);   castT(Wv,  Wvb);
  castT(Wqa, Wqab);  castT(Wka, Wkab);  castT(Wva, Wvab);
  castT(Wo,  Wob);   castT(Wao, Waob);

  // 2) projections (concat order: rows [0,512) = encoder, [512,2560) = image)
  gemm(hsb,  Wqb,  nullptr, Qf + (size_t)S_TXT * D, S_IMG);
  gemm(hsb,  Wkb,  nullptr, Kf + (size_t)S_TXT * D, S_IMG);
  gemm(hsb,  Wvb,  nullptr, Vf + (size_t)S_TXT * D, S_IMG);
  gemm(encb, Wqab, bqa,     Qf,                     S_TXT);
  gemm(encb, Wkab, bka,     Kf,                     S_TXT);
  gemm(encb, Wvab, bva,     Vf,                     S_TXT);

  // 3) RMSNorm + RoPE (Q, K) + bf16 pack; V is a straight cast
  {
    dim3 grid(H, S);
    rmsnorm_rope_kernel<<<grid, 128, 0, stream>>>(Qf, nqw, naq, cosb, sinb, Qbb, S_TXT);
    rmsnorm_rope_kernel<<<grid, 128, 0, stream>>>(Kf, nkw, nak, cosb, sinb, Kbb, S_TXT);
  }
  cast(Vf, Vbb, (long)S * D);

  // 4) attention
  {
    dim3 grid(S / 32, H);
    flash_attn_kernel<<<grid, 256, 0, stream>>>(Qbb, Kbb, Vbb, Of, S, H);
  }

  // 5) output projections straight into d_out (img first, then enc)
  cast(Of, Ob, (long)S * D);
  gemm(Ob + (size_t)S_TXT * D, Wob,  bo,  out,                     S_IMG);
  gemm(Ob,                     Waob, bao, out + (size_t)S_IMG * D, S_TXT);
}
